// RotarySelfAttention_35691178229915
// MI455X (gfx1250) — matrix operand order repaired and checked by reading, not compiled, not run on hardware
//
#include <hip/hip_runtime.h>
#include <hip/hip_bf16.h>

// CDNA5 / gfx1250, wave32. WMMA f16 -> f32 accumulate throughout.
// Async global->LDS copies (ASYNCcnt path) where the copy is transpose-free.

typedef __attribute__((ext_vector_type(16))) _Float16 v16h;
typedef __attribute__((ext_vector_type(8)))  float    v8f;
typedef __attribute__((ext_vector_type(4)))  _Float16 v4h;
typedef __attribute__((ext_vector_type(4)))  float    f32x4;
typedef __attribute__((ext_vector_type(4)))  int      v4i;

#define B_   8
#define T_   512
#define DM_  1024
#define H_   16
#define DH_  64
#define P_   3584
#define KTOT 4096   // P_ + T_

// Fragment union: 16 halfs (8 VGPRs) assembled from two 16-byte chunks.
union Frag {
  v16h  v;
  f32x4 f4[2];
  _Float16 h[16];
};

// ---- async global->LDS (gfx1250 GLOBAL_LOAD_ASYNC_TO_LDS_B128) ------------
#if defined(__gfx1250__) && __has_builtin(__builtin_amdgcn_global_load_async_to_lds_b128)
#define ASYNC_LDS 1
#else
#define ASYNC_LDS 0
#endif

__device__ __forceinline__ void copy16_to_lds(void* lds, const void* gsrc) {
#if ASYNC_LDS
  // Signature (from clang diagnostic): (v4i as(1)*, v4i as(3)*, imm off, imm cpol)
  __builtin_amdgcn_global_load_async_to_lds_b128(
      (__attribute__((address_space(1))) v4i*)gsrc,
      (__attribute__((address_space(3))) v4i*)lds,
      /*imm offset=*/0, /*cpol=*/0);
#else
  *(f32x4*)lds = *(const f32x4*)gsrc;
#endif
}

__device__ __forceinline__ void lds_copy_fence() {
#if ASYNC_LDS
  asm volatile("s_wait_asynccnt 0x0" ::: "memory");
#endif
}

// ---------------------------------------------------------------------------
// Kernel 1: fused QKV projection + bias + RoPE + new-token masking.
//   sel=0: Qf  (B,H,T,DH)   sel=1: Kc tail   sel=2: Vc tail
// Block: 128 threads (4 waves), 64x64 output tile; wave = 16x64 strip.
// ---------------------------------------------------------------------------
__global__ __launch_bounds__(128) void qkv_rope_kernel(
    const float* __restrict__ X,
    const float* __restrict__ Wq, const float* __restrict__ bq,
    const float* __restrict__ Wk, const float* __restrict__ bk,
    const float* __restrict__ Wv, const float* __restrict__ bv,
    const float* __restrict__ cosb, const float* __restrict__ sinb,
    const int*   __restrict__ ntc,
    _Float16* __restrict__ Qf, _Float16* __restrict__ Kc,
    _Float16* __restrict__ Vc)
{
  __shared__ __align__(16) _Float16 As[64][32];   // A tile, row-major [m][k]
  __shared__ __align__(16) _Float16 Bs[64][32];   // B tile, col-major [n][k]
  const int tid = threadIdx.x;
  const int w = tid >> 5, l = tid & 31;
  const int lane15 = l & 15, hi = l >> 4;
  const int n0 = blockIdx.x * 64, m0 = blockIdx.y * 64;
  const int sel = blockIdx.z;
  const float* Wm   = (sel == 0) ? Wq : (sel == 1) ? Wk : Wv;
  const float* bias = (sel == 0) ? bq : (sel == 1) ? bk : bv;

  v8f acc[4] = {};
  for (int k0 = 0; k0 < DM_; k0 += 32) {
    // A tile: 64x32 f32 -> f16 LDS (row-major); conversion forces VGPR path.
#pragma unroll
    for (int i = 0; i < 4; ++i) {
      int idx = tid + 128 * i;
      int row = idx >> 3, g = idx & 7;
      f32x4 xv = *(const f32x4*)(X + (size_t)(m0 + row) * DM_ + k0 + g * 4);
      v4h hv;
      hv.x = (_Float16)xv.x; hv.y = (_Float16)xv.y;
      hv.z = (_Float16)xv.z; hv.w = (_Float16)xv.w;
      *(v4h*)(&As[row][g * 4]) = hv;
    }
    // B tile: W[k][n] 32x64 f32 -> transposed f16 LDS [n][k]
#pragma unroll
    for (int i = 0; i < 4; ++i) {
      int idx = tid + 128 * i;
      int kr = idx >> 4, ng = idx & 15;
      f32x4 wv = *(const f32x4*)(Wm + (size_t)(k0 + kr) * DM_ + n0 + ng * 4);
      Bs[ng * 4 + 0][kr] = (_Float16)wv.x;
      Bs[ng * 4 + 1][kr] = (_Float16)wv.y;
      Bs[ng * 4 + 2][kr] = (_Float16)wv.z;
      Bs[ng * 4 + 3][kr] = (_Float16)wv.w;
    }
    __syncthreads();

    // A fragment (16x32 CDNA5 layout): row = lane&15, K chunks hi*8, hi*8+16
    Frag af;
    const int am  = w * 16 + lane15;
    const int kb8 = hi * 8;
    af.f4[0] = *(const f32x4*)(&As[am][kb8]);
    af.f4[1] = *(const f32x4*)(&As[am][kb8 + 16]);
#pragma unroll
    for (int j = 0; j < 4; ++j) {
      // B fragment: lane = column, K chunks hi*8, hi*8+16
      Frag bf;
      const int bn = j * 16 + lane15;
      const int ko = hi * 8;
      bf.f4[0] = *(const f32x4*)(&Bs[bn][ko]);
      bf.f4[1] = *(const f32x4*)(&Bs[bn][ko + 16]);
      acc[j] = __builtin_amdgcn_wmma_f32_16x16x32_f16(
          false, af.v, false, bf.v, (short)0, acc[j], false, false);
    }
    __syncthreads();
  }

  // Epilogue: bias, rope (q/k), mask, store f16.
  float vals[4][8];
#pragma unroll
  for (int j = 0; j < 4; ++j) {
    const float bj = bias[n0 + j * 16 + lane15];
#pragma unroll
    for (int r = 0; r < 8; ++r) vals[j][r] = acc[j][r] + bj;
  }
  const int hh = n0 >> 6;  // head index (64-wide N tiles align to heads)
#pragma unroll
  for (int r = 0; r < 8; ++r) {
    const int row = m0 + w * 16 + r + hi * 8;   // C layout: M = r + hi*8
    const int b = row >> 9, t = row & 511;
    const float msk = (t < ntc[b]) ? 1.0f : 0.0f;
#pragma unroll
    for (int j = 0; j < 4; ++j) {
      const int d = j * 16 + lane15;
      float outv;
      if (sel < 2) {
        // rotate_half: d<32 -> -x[d+32] (acc j+2); d>=32 -> +x[d-32] (acc j-2)
        const float rot = (j < 2) ? -vals[j + 2][r] : vals[j - 2][r];
        outv = vals[j][r] * cosb[t * DH_ + d] + rot * sinb[t * DH_ + d];
      } else {
        outv = vals[j][r];
      }
      outv *= msk;
      const _Float16 hv = (_Float16)outv;
      if (sel == 0) {
        Qf[(((size_t)b * H_ + hh) * T_ + t) * DH_ + d] = hv;
      } else {
        _Float16* dst = (sel == 1) ? Kc : Vc;
        dst[(((size_t)b * H_ + hh) * KTOT + P_ + t) * DH_ + d] = hv;
      }
    }
  }
}

// ---------------------------------------------------------------------------
// Kernel 2: masked f32->f16 repack of past K/V into combined caches [0,P).
// ---------------------------------------------------------------------------
__global__ void past_pack_kernel(
    const float* __restrict__ pk, const float* __restrict__ pv,
    const int* __restrict__ plen,
    _Float16* __restrict__ Kc, _Float16* __restrict__ Vc)
{
  const size_t per = (size_t)B_ * H_ * P_ * (DH_ / 4);  // float4 units per array
  size_t gid = (size_t)blockIdx.x * blockDim.x + threadIdx.x;
  if (gid >= 2 * per) return;
  const int which = (gid >= per);
  const size_t e = which ? gid - per : gid;
  const int d4 = (int)(e & 15);
  size_t tmp = e >> 4;
  const int p = (int)(tmp % P_);
  tmp /= P_;
  const int h = (int)(tmp & 15);
  const int b = (int)(tmp >> 4);
  int tp = plen[b]; tp = tp < 0 ? 0 : (tp > P_ ? P_ : tp);
  const float m = (p < tp) ? 1.0f : 0.0f;
  const float* src = which ? pv : pk;
  f32x4 xv = *(const f32x4*)(src + e * 4);
  v4h hv;
  hv.x = (_Float16)(xv.x * m); hv.y = (_Float16)(xv.y * m);
  hv.z = (_Float16)(xv.z * m); hv.w = (_Float16)(xv.w * m);
  _Float16* dst = which ? Vc : Kc;
  *(v4h*)(dst + (((size_t)b * H_ + h) * KTOT + p) * DH_ + d4 * 4) = hv;
}

// ---------------------------------------------------------------------------
// Kernel 3: flash attention. Block = (b,h, 64 q rows), 4 waves x 16 q rows.
// K blocks streamed via async global->LDS; online softmax; WMMA both GEMMs.
// ---------------------------------------------------------------------------
__global__ __launch_bounds__(128) void flash_attn_kernel(
    const _Float16* __restrict__ Qf, const _Float16* __restrict__ Kc,
    const _Float16* __restrict__ Vc,
    const int* __restrict__ plen, const int* __restrict__ ntc,
    _Float16* __restrict__ Oat)
{
  __shared__ __align__(16) _Float16 Ks[64][64];      // [key][d] row-major
  __shared__ __align__(16) _Float16 Vt[64][64];      // [d][key] transposed
  __shared__ __align__(16) _Float16 Ps[4][16][64];   // per-wave P tile [m][key]
  const int tid = threadIdx.x;
  const int w = tid >> 5, l = tid & 31;
  const int lane15 = l & 15, hi = l >> 4, kb8 = hi * 8;
  const int bh = blockIdx.y;
  const int b = bh >> 4;
  const int q0 = blockIdx.x * 64;
  int tp = plen[b]; tp = tp < 0 ? 0 : (tp > P_ ? P_ : tp);
  const int tot = tp + ntc[b];                       // valid iff key < tot
  const _Float16* Qb = Qf + (size_t)bh * T_ * DH_;
  const _Float16* Kb = Kc + (size_t)bh * KTOT * DH_;
  const _Float16* Vb = Vc + (size_t)bh * KTOT * DH_;

  // Q fragments for both 32-wide d chunks, loaded once.
  Frag qf[2];
  {
    const _Float16* qr = Qb + (size_t)(q0 + w * 16 + lane15) * DH_;
    qf[0].f4[0] = *(const f32x4*)(qr + kb8);
    qf[0].f4[1] = *(const f32x4*)(qr + kb8 + 16);
    qf[1].f4[0] = *(const f32x4*)(qr + 32 + kb8);
    qf[1].f4[1] = *(const f32x4*)(qr + 32 + kb8 + 16);
  }

  v8f O[4] = {};
  float Mrow[8], Lrow[8];
#pragma unroll
  for (int r = 0; r < 8; ++r) { Mrow[r] = -1.0e30f; Lrow[r] = 0.0f; }
  const float scale = 0.125f;  // 1/sqrt(64)

  for (int kb = 0; kb < KTOT / 64; ++kb) {
    const int kstart = kb * 64;
#pragma unroll
    for (int i = 0; i < 4; ++i) {  // K block: async 16B global->LDS copies
      int idx = tid + 128 * i;
      int key = idx >> 3, g = idx & 7;
      copy16_to_lds(&Ks[key][g * 8],
                    Kb + (size_t)(kstart + key) * DH_ + g * 8);
    }
#pragma unroll
    for (int i = 0; i < 4; ++i) {  // V block: transpose into [d][key]
      int idx = tid + 128 * i;
      int key = idx >> 3, g = idx & 7;
      union { f32x4 f; _Float16 hh[8]; } u;
      u.f = *(const f32x4*)(Vb + (size_t)(kstart + key) * DH_ + g * 8);
#pragma unroll
      for (int c = 0; c < 8; ++c) Vt[g * 8 + c][key] = u.hh[c];
    }
    lds_copy_fence();     // own-wave s_wait_asynccnt 0
    __syncthreads();      // cross-wave visibility

    // Prefetch next K/V block (overlaps the WMMA work below).
    if (kb + 1 < KTOT / 64) {
      const size_t nrow = (size_t)(kstart + 64 + (tid >> 1)) * DH_;
      __builtin_prefetch(Kb + nrow, 0, 1);
      __builtin_prefetch(Vb + nrow, 0, 1);
    }

    // Scores S = Q * K^T  (K-dim = d = 64 -> 2 wmma per 16x16 tile)
    v8f S[4] = {};
#pragma unroll
    for (int j = 0; j < 4; ++j) {
      const int keyl = j * 16 + lane15;
#pragma unroll
      for (int kc = 0; kc < 2; ++kc) {
        Frag bf;
        const int ko = kc * 32 + hi * 8;
        bf.f4[0] = *(const f32x4*)(&Ks[keyl][ko]);
        bf.f4[1] = *(const f32x4*)(&Ks[keyl][ko + 16]);
        S[j] = __builtin_amdgcn_wmma_f32_16x16x32_f16(
            false, qf[kc].v, false, bf.v, (short)0, S[j], false, false);
      }
    }
    // Scale + mask (exactly: key_index < total_lengths)
#pragma unroll
    for (int j = 0; j < 4; ++j) {
      const int kg = kstart + j * 16 + lane15;
      const bool valid = kg < tot;
#pragma unroll
      for (int r = 0; r < 8; ++r)
        S[j][r] = valid ? S[j][r] * scale : -1.0e9f;
    }
    // Online softmax: per-row reductions across the 16-lane half-wave.
#pragma unroll
    for (int r = 0; r < 8; ++r) {
      float mx = fmaxf(fmaxf(S[0][r], S[1][r]), fmaxf(S[2][r], S[3][r]));
#pragma unroll
      for (int s = 8; s; s >>= 1) mx = fmaxf(mx, __shfl_xor(mx, s));
      const float Mnew = fmaxf(Mrow[r], mx);
      const float alpha = __expf(Mrow[r] - Mnew);
      float rsum = 0.0f;
#pragma unroll
      for (int j = 0; j < 4; ++j) {
        const float p = __expf(S[j][r] - Mnew);
        S[j][r] = p;
        rsum += p;
      }
#pragma unroll
      for (int s = 8; s; s >>= 1) rsum += __shfl_xor(rsum, s);
      Lrow[r] = Lrow[r] * alpha + rsum;
      Mrow[r] = Mnew;
#pragma unroll
      for (int j = 0; j < 4; ++j) O[j][r] *= alpha;
    }
    // P (C layout) -> per-wave LDS patch -> A-layout fragments.
#pragma unroll
    for (int r = 0; r < 8; ++r)
#pragma unroll
      for (int j = 0; j < 4; ++j)
        Ps[w][r + hi * 8][j * 16 + lane15] = (_Float16)S[j][r];
    Frag pf[2];
#pragma unroll
    for (int kc = 0; kc < 2; ++kc) {
      pf[kc].f4[0] = *(const f32x4*)(&Ps[w][lane15][kc * 32 + kb8]);
      pf[kc].f4[1] = *(const f32x4*)(&Ps[w][lane15][kc * 32 + kb8 + 16]);
    }
    // O += P * V  (K-dim = key = 64 -> 2 wmma per 16x16 output tile)
#pragma unroll
    for (int j = 0; j < 4; ++j) {
      const int dl = j * 16 + lane15;
#pragma unroll
      for (int kc = 0; kc < 2; ++kc) {
        Frag vf;
        const int ko = kc * 32 + hi * 8;
        vf.f4[0] = *(const f32x4*)(&Vt[dl][ko]);
        vf.f4[1] = *(const f32x4*)(&Vt[dl][ko + 16]);
        O[j] = __builtin_amdgcn_wmma_f32_16x16x32_f16(
            false, pf[kc].v, false, vf.v, (short)0, O[j], false, false);
      }
    }
    __syncthreads();
  }

  // Normalize and store f16 at (b*T + t, h*64 + d) of the (4096x1024) matrix.
  const int h = bh & 15;
#pragma unroll
  for (int r = 0; r < 8; ++r) {
    const float inv = 1.0f / Lrow[r];
    const int t = q0 + w * 16 + r + hi * 8;
#pragma unroll
    for (int j = 0; j < 4; ++j) {
      const int col = h * DH_ + j * 16 + lane15;
      Oat[((size_t)b * T_ + t) * DM_ + col] = (_Float16)(O[j][r] * inv);
    }
  }
}

// ---------------------------------------------------------------------------
// Kernel 4: output projection. Oat(f16, 4096x1024) x Wo + bo, row-masked, f32.
// A tile staged via async global->LDS (already f16, straight copy).
// ---------------------------------------------------------------------------
__global__ __launch_bounds__(128) void out_proj_kernel(
    const _Float16* __restrict__ Oat, const float* __restrict__ Wo,
    const float* __restrict__ bo, const int* __restrict__ ntc,
    float* __restrict__ Out)
{
  __shared__ __align__(16) _Float16 As[64][32];
  __shared__ __align__(16) _Float16 Bs[64][32];
  const int tid = threadIdx.x;
  const int w = tid >> 5, l = tid & 31;
  const int lane15 = l & 15, hi = l >> 4;
  const int n0 = blockIdx.x * 64, m0 = blockIdx.y * 64;

  v8f acc[4] = {};
  for (int k0 = 0; k0 < DM_; k0 += 32) {
#pragma unroll
    for (int i = 0; i < 2; ++i) {  // A tile already f16: async copies
      int idx = tid + 128 * i;
      int row = idx >> 2, g = idx & 3;
      copy16_to_lds(&As[row][g * 8],
                    Oat + (size_t)(m0 + row) * DM_ + k0 + g * 8);
    }
#pragma unroll
    for (int i = 0; i < 4; ++i) {  // Wo: f32 -> f16, transposed
      int idx = tid + 128 * i;
      int kr = idx >> 4, ng = idx & 15;
      f32x4 wv = *(const f32x4*)(Wo + (size_t)(k0 + kr) * DM_ + n0 + ng * 4);
      Bs[ng * 4 + 0][kr] = (_Float16)wv.x;
      Bs[ng * 4 + 1][kr] = (_Float16)wv.y;
      Bs[ng * 4 + 2][kr] = (_Float16)wv.z;
      Bs[ng * 4 + 3][kr] = (_Float16)wv.w;
    }
    lds_copy_fence();
    __syncthreads();

    Frag af;
    const int am  = w * 16 + lane15;
    const int kb8 = hi * 8;
    af.f4[0] = *(const f32x4*)(&As[am][kb8]);
    af.f4[1] = *(const f32x4*)(&As[am][kb8 + 16]);
#pragma unroll
    for (int j = 0; j < 4; ++j) {
      Frag bf;
      const int bn = j * 16 + lane15;
      const int ko = hi * 8;
      bf.f4[0] = *(const f32x4*)(&Bs[bn][ko]);
      bf.f4[1] = *(const f32x4*)(&Bs[bn][ko + 16]);
      acc[j] = __builtin_amdgcn_wmma_f32_16x16x32_f16(
          false, af.v, false, bf.v, (short)0, acc[j], false, false);
    }
    __syncthreads();
  }

#pragma unroll
  for (int r = 0; r < 8; ++r) {
    const int row = m0 + w * 16 + r + hi * 8;
    const int b = row >> 9, t = row & 511;
    const float msk = (t < ntc[b]) ? 1.0f : 0.0f;
#pragma unroll
    for (int j = 0; j < 4; ++j) {
      const int n = n0 + j * 16 + lane15;
      Out[(size_t)row * DM_ + n] = (acc[j][r] + bo[n]) * msk;
    }
  }
}

// ---------------------------------------------------------------------------
extern "C" void kernel_launch(void* const* d_in, const int* in_sizes, int n_in,
                              void* d_out, int out_size, void* d_ws, size_t ws_size,
                              hipStream_t stream) {
  (void)in_sizes; (void)n_in; (void)out_size; (void)ws_size;
  const float* x_new  = (const float*)d_in[0];
  const float* cosb   = (const float*)d_in[1];   // (1,1,T,DH)
  const float* sinb   = (const float*)d_in[2];
  const float* past_k = (const float*)d_in[3];
  const float* past_v = (const float*)d_in[4];
  const int*   plen   = (const int*)d_in[5];
  /* d_in[6] = valid_new_mask (bool) — recomputed from new_token_counts */
  const int*   ntc    = (const int*)d_in[7];
  const float* Wq = (const float*)d_in[8];
  const float* bq = (const float*)d_in[9];
  const float* Wk = (const float*)d_in[10];
  const float* bk = (const float*)d_in[11];
  const float* Wv = (const float*)d_in[12];
  const float* bv = (const float*)d_in[13];
  const float* Wo = (const float*)d_in[14];
  const float* bo = (const float*)d_in[15];
  float* Out = (float*)d_out;

  // Workspace layout (~144 MB total):
  char* ws = (char*)d_ws;
  const size_t qBytes  = (size_t)B_ * H_ * T_ * DH_ * sizeof(_Float16);   //  8 MB
  const size_t kvBytes = (size_t)B_ * H_ * KTOT * DH_ * sizeof(_Float16); // 64 MB
  _Float16* Qf  = (_Float16*)ws;
  _Float16* Kc  = (_Float16*)(ws + qBytes);
  _Float16* Vc  = (_Float16*)(ws + qBytes + kvBytes);
  _Float16* Oat = (_Float16*)(ws + qBytes + 2 * kvBytes);                 //  8 MB

  qkv_rope_kernel<<<dim3(DM_ / 64, (B_ * T_) / 64, 3), 128, 0, stream>>>(
      x_new, Wq, bq, Wk, bk, Wv, bv, cosb, sinb, ntc, Qf, Kc, Vc);

  const size_t tot4 = 2ull * B_ * H_ * P_ * (DH_ / 4);
  past_pack_kernel<<<dim3((unsigned)((tot4 + 255) / 256)), 256, 0, stream>>>(
      past_k, past_v, plen, Kc, Vc);

  flash_attn_kernel<<<dim3(T_ / 64, B_ * H_), 128, 0, stream>>>(
      Qf, Kc, Vc, plen, ntc, Oat);

  out_proj_kernel<<<dim3(DM_ / 64, (B_ * T_) / 64), 128, 0, stream>>>(
      Oat, Wo, bo, ntc, Out);
}
